// LSTMTemporalEncoder_28415503630973
// MI455X (gfx1250) — matrix operand order repaired and checked by reading, not compiled, not run on hardware
//
#include <hip/hip_runtime.h>

// ---------------------------------------------------------------------------
// Fused 2-layer LSTM (N=8192, T=64, D=2, H=256) + projection(E=64) + LayerNorm
// CDNA5 / gfx1250: bf16 WMMA (v_wmma_f32_16x16x32_bf16), wave32, WGP LDS.
// ---------------------------------------------------------------------------

typedef __attribute__((ext_vector_type(16))) __bf16 v16bf;
typedef __attribute__((ext_vector_type(8)))  float  v8f;

union Frag {
    v16bf v;
    uint4 q[2];
    unsigned int u[8];
};

#define NBATCH 8192
#define TSTEPS 64
#define DIN    2
#define HDIM   256
#define EDIM   64
#define G4     1024   // 4*H

// workspace layout (dword offsets)
#define OFF_WHH0 0          // 8kt*64c*32lane*8dw = 131072 dwords (bf16-packed)
#define OFF_WIH1 131072
#define OFF_WHH1 262144
#define OFF_WP   393216     // 8kt*4c*32lane*8dw = 8192 dwords
#define OFF_B0   401408     // 1024 floats (bih0+bhh0)
#define OFF_B1   402432     // 1024 floats (bih1+bhh1)

__device__ __forceinline__ unsigned short f2bf(float f) {
    unsigned int u = __float_as_uint(f);
    u += 0x7FFFu + ((u >> 16) & 1u);           // round-to-nearest-even
    return (unsigned short)(u >> 16);
}
__device__ __forceinline__ float sigm(float x)      { return 1.0f / (1.0f + __expf(-x)); }
__device__ __forceinline__ float tanh_fast(float x) { return 1.0f - 2.0f / (__expf(2.0f * x) + 1.0f); }

// ---------------------------------------------------------------------------
// Prep: swizzle fp32 weight W[j, k] (j = output/gate dim, k = K dim) into the
// bf16 B-operand tile layout: [kt][coltile][lane][8 dwords].
// Lane layout per 32x16 B tile: lanes 0-15 hold K=0..15 (2 per dword),
// lanes 16-31 hold K=16..31; N = coltile*16 + (lane&15).
// ---------------------------------------------------------------------------
__global__ void swizzle_w(const float* __restrict__ W, unsigned int* __restrict__ dst,
                          int Kdim, int ncoltiles, int total) {
    int idx = blockIdx.x * blockDim.x + threadIdx.x;
    if (idx >= total) return;
    int v    = idx & 7;
    int lane = (idx >> 3) & 31;
    int c    = (idx >> 8) % ncoltiles;
    int kt   = (idx >> 8) / ncoltiles;
    int j = c * 16 + (lane & 15);
    int k = kt * 32 + (lane >> 4) * 16 + v * 2;
    unsigned int lo = f2bf(W[j * Kdim + k]);
    unsigned int hi = f2bf(W[j * Kdim + k + 1]);
    dst[idx] = lo | (hi << 16);
}

__global__ void add_bias(const float* __restrict__ a, const float* __restrict__ b,
                         float* __restrict__ d, int n) {
    int i = blockIdx.x * blockDim.x + threadIdx.x;
    if (i < n) d[i] = a[i] + b[i];
}

// ---------------------------------------------------------------------------
// One K-pass: acc[gate][cl][mt] += A(h tile from LDS) x B(swizzled weights).
// A-layout used here (contiguous k, same order as the B swizzle): lane m=(l&15); lanes<16 take K {0..15},
// lanes>=16 take K {16..31} -> two b128 LDS loads per (mt,kt).
// ---------------------------------------------------------------------------
static __device__ __forceinline__ void kpass(const unsigned short* hlds,
                                             const unsigned int* __restrict__ Bsw,
                                             int w, int l, v8f acc[4][2][2]) {
    const int nloc = l & 15, hsel = l >> 4;
#pragma unroll
    for (int kt = 0; kt < 8; ++kt) {
        Frag A[2];
#pragma unroll
        for (int mt = 0; mt < 2; ++mt) {
            int m  = mt * 16 + nloc;
            int k0 = kt * 32 + hsel * 16;
            A[mt].q[0] = *reinterpret_cast<const uint4*>(&hlds[m * HDIM + k0]);
            A[mt].q[1] = *reinterpret_cast<const uint4*>(&hlds[m * HDIM + k0 + 8]);
        }
#pragma unroll
        for (int gt = 0; gt < 4; ++gt) {
#pragma unroll
            for (int cl = 0; cl < 2; ++cl) {
                int c = gt * 16 + w * 2 + cl;      // global coltile (j = 16*c)
                Frag B;
                const uint4* bp4 = reinterpret_cast<const uint4*>(Bsw)
                                   + (((kt * 64 + c) * 32) + l) * 2;
                B.q[0] = bp4[0];
                B.q[1] = bp4[1];
#pragma unroll
                for (int mt = 0; mt < 2; ++mt) {
                    acc[gt][cl][mt] = __builtin_amdgcn_wmma_f32_16x16x32_bf16(
                        false, A[mt].v, false, B.v, (short)0, acc[gt][cl][mt],
                        false, false);
                }
            }
        }
    }
}

// Cell update + bf16 h store to LDS (caller guarantees barrier before/after).
static __device__ __forceinline__ void cell_update(v8f acc[4][2][2], v8f cst[2][2],
                                                   unsigned short* hlds, int w, int l) {
    const int nloc = l & 15, hsel = l >> 4;
#pragma unroll
    for (int cl = 0; cl < 2; ++cl) {
#pragma unroll
        for (int mt = 0; mt < 2; ++mt) {
#pragma unroll
            for (int r = 0; r < 8; ++r) {
                float iv = sigm(acc[0][cl][mt][r]);
                float fv = sigm(acc[1][cl][mt][r]);
                float gv = tanh_fast(acc[2][cl][mt][r]);
                float ov = sigm(acc[3][cl][mt][r]);
                float cc = fv * cst[cl][mt][r] + iv * gv;
                cst[cl][mt][r] = cc;
                float hv = ov * tanh_fast(cc);
                int m = mt * 16 + r + 8 * hsel;              // D-layout row
                int k = w * 32 + cl * 16 + nloc;             // hidden column
                hlds[m * HDIM + k] = f2bf(hv);
            }
        }
    }
}

// ---------------------------------------------------------------------------
// Main fused kernel: one workgroup = 32 batch rows, 8 waves.
// ---------------------------------------------------------------------------
__global__ __launch_bounds__(256)
void lstm2_fused(const float* __restrict__ x, const float* __restrict__ Wih0,
                 const unsigned int* __restrict__ Bwhh0,
                 const unsigned int* __restrict__ Bwih1,
                 const unsigned int* __restrict__ Bwhh1,
                 const unsigned int* __restrict__ Bwp,
                 const float* __restrict__ b0, const float* __restrict__ b1,
                 const float* __restrict__ bp, const float* __restrict__ gma,
                 const float* __restrict__ bta, float* __restrict__ out) {
    __shared__ float bias0[G4];
    __shared__ float bias1[G4];
    __shared__ float wih0s[G4 * DIN];
    __shared__ float xs[32 * DIN];
    __shared__ float zs[32 * EDIM];
    __shared__ unsigned short h0s[32 * HDIM];
    __shared__ unsigned short h1s[32 * HDIM];

    const int tid  = threadIdx.x;
    const int w    = tid >> 5;
    const int l    = tid & 31;
    const int nloc = l & 15;
    const int hsel = l >> 4;
    const int n0   = blockIdx.x * 32;

    for (int i = tid; i < G4; i += 256) { bias0[i] = b0[i]; bias1[i] = b1[i]; }
    for (int i = tid; i < G4 * DIN; i += 256) wih0s[i] = Wih0[i];
    for (int i = tid; i < 32 * HDIM; i += 256) { h0s[i] = 0; h1s[i] = 0; }

    v8f c0[2][2], c1[2][2];
#pragma unroll
    for (int cl = 0; cl < 2; ++cl)
#pragma unroll
        for (int mt = 0; mt < 2; ++mt)
#pragma unroll
            for (int r = 0; r < 8; ++r) { c0[cl][mt][r] = 0.0f; c1[cl][mt][r] = 0.0f; }

    __syncthreads();

#pragma unroll 1
    for (int t = 0; t < TSTEPS; ++t) {
        // stage x_t tile (32 rows x 2)
        if (tid < 64) xs[tid] = x[(n0 + (tid >> 1)) * (TSTEPS * DIN) + t * DIN + (tid & 1)];
        __syncthreads();

        v8f acc[4][2][2];

        // ---- layer 0: init with bias + tiny input projection (D=2) ----
#pragma unroll
        for (int gt = 0; gt < 4; ++gt) {
#pragma unroll
            for (int cl = 0; cl < 2; ++cl) {
                int j = gt * 256 + w * 32 + cl * 16 + nloc;
                float bj = bias0[j];
                float w0v = wih0s[2 * j + 0], w1v = wih0s[2 * j + 1];
#pragma unroll
                for (int mt = 0; mt < 2; ++mt)
#pragma unroll
                    for (int r = 0; r < 8; ++r) {
                        int m = mt * 16 + r + 8 * hsel;
                        acc[gt][cl][mt][r] = bj + xs[2 * m] * w0v + xs[2 * m + 1] * w1v;
                    }
            }
        }
        kpass(h0s, Bwhh0, w, l, acc);          // + h0_{t-1} @ Whh0^T
        __syncthreads();                        // all reads of h0s done
        cell_update(acc, c0, h0s, w, l);        // writes new h0 (bf16) to LDS
        __syncthreads();

        // ---- layer 1: g = h0_t @ Wih1^T + h1_{t-1} @ Whh1^T + b ----
#pragma unroll
        for (int gt = 0; gt < 4; ++gt)
#pragma unroll
            for (int cl = 0; cl < 2; ++cl) {
                float bj = bias1[gt * 256 + w * 32 + cl * 16 + nloc];
#pragma unroll
                for (int mt = 0; mt < 2; ++mt)
#pragma unroll
                    for (int r = 0; r < 8; ++r) acc[gt][cl][mt][r] = bj;
            }
        kpass(h0s, Bwih1, w, l, acc);
        kpass(h1s, Bwhh1, w, l, acc);
        __syncthreads();                        // all reads of h1s done
        cell_update(acc, c1, h1s, w, l);        // writes new h1 (bf16) to LDS
        __syncthreads();
    }

    // ---- projection z = h1_T @ Wp^T + bp  (wave 0, WMMA, E=64) ----
    if (w == 0) {
        v8f pz[4][2];
#pragma unroll
        for (int ct = 0; ct < 4; ++ct) {
            float bj = bp[ct * 16 + nloc];
#pragma unroll
            for (int mt = 0; mt < 2; ++mt)
#pragma unroll
                for (int r = 0; r < 8; ++r) pz[ct][mt][r] = bj;
        }
#pragma unroll
        for (int kt = 0; kt < 8; ++kt) {
            Frag A[2];
#pragma unroll
            for (int mt = 0; mt < 2; ++mt) {
                int m  = mt * 16 + nloc;
                int k0 = kt * 32 + hsel * 16;
                A[mt].q[0] = *reinterpret_cast<const uint4*>(&h1s[m * HDIM + k0]);
                A[mt].q[1] = *reinterpret_cast<const uint4*>(&h1s[m * HDIM + k0 + 8]);
            }
#pragma unroll
            for (int ct = 0; ct < 4; ++ct) {
                Frag B;
                const uint4* bp4 = reinterpret_cast<const uint4*>(Bwp)
                                   + (((kt * 4 + ct) * 32) + l) * 2;
                B.q[0] = bp4[0];
                B.q[1] = bp4[1];
#pragma unroll
                for (int mt = 0; mt < 2; ++mt)
                    pz[ct][mt] = __builtin_amdgcn_wmma_f32_16x16x32_bf16(
                        false, A[mt].v, false, B.v, (short)0, pz[ct][mt], false, false);
            }
        }
#pragma unroll
        for (int ct = 0; ct < 4; ++ct)
#pragma unroll
            for (int mt = 0; mt < 2; ++mt)
#pragma unroll
                for (int r = 0; r < 8; ++r)
                    zs[(mt * 16 + r + 8 * hsel) * EDIM + ct * 16 + nloc] = pz[ct][mt][r];
    }
    __syncthreads();

    // ---- LayerNorm over E=64, one row per thread ----
    if (tid < 32) {
        const float* zr = &zs[tid * EDIM];
        float mu = 0.0f;
        for (int j = 0; j < EDIM; ++j) mu += zr[j];
        mu *= (1.0f / EDIM);
        float var = 0.0f;
        for (int j = 0; j < EDIM; ++j) { float d = zr[j] - mu; var += d * d; }
        var *= (1.0f / EDIM);
        float rs = rsqrtf(var + 1e-5f);
        float* orow = &out[(n0 + tid) * EDIM];
        for (int j = 0; j < EDIM; ++j)
            orow[j] = (zr[j] - mu) * rs * gma[j] + bta[j];
    }
}

// ---------------------------------------------------------------------------
extern "C" void kernel_launch(void* const* d_in, const int* in_sizes, int n_in,
                              void* d_out, int out_size, void* d_ws, size_t ws_size,
                              hipStream_t stream) {
    const float* x    = (const float*)d_in[0];
    const float* Wih0 = (const float*)d_in[1];
    const float* Whh0 = (const float*)d_in[2];
    const float* bih0 = (const float*)d_in[3];
    const float* bhh0 = (const float*)d_in[4];
    const float* Wih1 = (const float*)d_in[5];
    const float* Whh1 = (const float*)d_in[6];
    const float* bih1 = (const float*)d_in[7];
    const float* bhh1 = (const float*)d_in[8];
    const float* Wp   = (const float*)d_in[9];
    const float* bp   = (const float*)d_in[10];
    const float* gma  = (const float*)d_in[11];
    const float* bta  = (const float*)d_in[12];
    float* out = (float*)d_out;

    unsigned int* ws = (unsigned int*)d_ws;

    // weight swizzle + bias fold (cheap, stream-ordered)
    const int totHH = 8 * 64 * 32 * 8;   // 131072 dwords per 1024x256 matrix
    const int totP  = 8 * 4 * 32 * 8;    // 8192 dwords for 64x256 Wp
    swizzle_w<<<(totHH + 255) / 256, 256, 0, stream>>>(Whh0, ws + OFF_WHH0, HDIM, 64, totHH);
    swizzle_w<<<(totHH + 255) / 256, 256, 0, stream>>>(Wih1, ws + OFF_WIH1, HDIM, 64, totHH);
    swizzle_w<<<(totHH + 255) / 256, 256, 0, stream>>>(Whh1, ws + OFF_WHH1, HDIM, 64, totHH);
    swizzle_w<<<(totP + 255) / 256, 256, 0, stream>>>(Wp, ws + OFF_WP, HDIM, 4, totP);
    add_bias<<<4, 256, 0, stream>>>(bih0, bhh0, (float*)(ws + OFF_B0), G4);
    add_bias<<<4, 256, 0, stream>>>(bih1, bhh1, (float*)(ws + OFF_B1), G4);

    // fused 2-layer LSTM + projection + LayerNorm: 256 WGs x 32 rows
    lstm2_fused<<<NBATCH / 32, 256, 0, stream>>>(
        x, Wih0,
        ws + OFF_WHH0, ws + OFF_WIH1, ws + OFF_WHH1, ws + OFF_WP,
        (const float*)(ws + OFF_B0), (const float*)(ws + OFF_B1),
        bp, gma, bta, out);
}
